// DDGNet_88854283420133
// MI455X (gfx1250) — matrix operand order repaired and checked by reading, not compiled, not run on hardware
//
#include <hip/hip_runtime.h>
#include <hip/hip_bf16.h>
#include <math.h>

// ---------------------------------------------------------------------------
// CDNA5 (gfx1250) DDGNet pipeline, round 3.
// - bf16 NHWC activations, weights pre-packed to WMMA A-fragment layout.
// - Convs + 32->441 projection: implicit GEMM with 2x2 register blocking:
//   one wave = 32 out-channels x 32 pixels, 4 accumulators, 4 WMMAs per
//   (tap, chunk) on 2 A + 2 B fragment loads (halves fragment traffic/MAC,
//   gives 4 independent WMMAs to overlap loads and hazard windows).
// - softmax/blur on [pixel][448] contiguous rows (L2-resident 117MB tensor).
// ---------------------------------------------------------------------------

typedef __attribute__((ext_vector_type(16))) __bf16 v16bf;
typedef __attribute__((ext_vector_type(8)))  __bf16 v8bf;
typedef __attribute__((ext_vector_type(8)))  float  v8f;

__device__ __forceinline__ float lrelu(float x) { return x > 0.f ? x : 0.1f * x; }

// ---------------------------------------------------------------------------
// Weight packing: OIHW fp32 -> bf16 A-fragment layout
//   pw[(((mt*ntap + tap)*nch + cc)*32 + lane)*16 + h]
//   k(h,lane) = h + ((lane<16)?0:16), M = lane%16 (contiguous k, matches B loads)
// ---------------------------------------------------------------------------
__global__ void packw(const float* __restrict__ wt, __bf16* __restrict__ pw,
                      int Cout, int CinSrc, int CinPad, int ntap, int total)
{
  int idx = blockIdx.x * 256 + threadIdx.x;
  if (idx >= total) return;
  const int nch = CinPad >> 5;
  int h = idx & 15; int t = idx >> 4;
  int lane = t & 31; t >>= 5;
  int cc = t % nch; t /= nch;
  int tap = t % ntap; int mt = t / ntap;
  int k  = h + ((lane < 16) ? 0 : 16);
  int c  = cc * 32 + k;
  int oc = mt * 16 + (lane & 15);
  float wv = 0.f;
  if (oc < Cout && c < CinSrc)
    wv = wt[((size_t)oc * CinSrc + c) * ntap + tap];
  pw[idx] = (__bf16)wv;
}

// ---------------------------------------------------------------------------
// Pad fp32 NCHW (3,256,256) image -> bf16 NHWC (65536, 32), ch 3..31 = 0.
// ---------------------------------------------------------------------------
__global__ void img_to_bf16(const float* __restrict__ img, __bf16* __restrict__ out)
{
  int idx = blockIdx.x * 256 + threadIdx.x;
  if (idx >= 65536 * 32) return;
  const int c = idx & 31, p = idx >> 5;
  out[idx] = (__bf16)((c < 3) ? img[c * 65536 + p] : 0.f);
}

// ---------------------------------------------------------------------------
// 3x3 conv (pad=1) implicit GEMM, 2x2 register-blocked WMMA bf16.
// Requires: Cin%32==0, Cout%32==0, (Ho*Wo)%32==0.
// Optional concat: c < C1 -> inA[p*C1+c], else inB[p*C2+(c-C1)]; 16-channel
// segments never straddle C1 (all splits multiples of 32).
// ---------------------------------------------------------------------------
__global__ __launch_bounds__(128)
void conv3x3_wmma(const __bf16* __restrict__ inA, const __bf16* __restrict__ inB,
                  int C1, int C2,
                  const v16bf* __restrict__ pw, const float* __restrict__ bias,
                  __bf16* __restrict__ out, int Cout,
                  int Hi, int Wi, int Wo,
                  int stride, int relu, int MtB, int tiles)
{
  const int lane = threadIdx.x & 31;
  const int tile = blockIdx.x * 4 + (threadIdx.x >> 5);
  if (tile >= tiles) return;                       // wave-uniform exit
  const int mt2 = tile % MtB, nt2 = tile / MtB;    // block shares pixel tiles
  const int mt = mt2 * 2;
  const int nch = (C1 + C2) >> 5;

  const int p0 = nt2 * 32 + (lane & 15);
  const int p1 = p0 + 16;
  const int oy0 = p0 / Wo, ox0 = p0 % Wo;
  const int oy1 = p1 / Wo, ox1 = p1 % Wo;
  const int kbofs = (lane < 16) ? 0 : 16;

  v8f acc00 = {}, acc01 = {}, acc10 = {}, acc11 = {};

  const v16bf* pw0 = pw + ((size_t)mt * 9 * nch) * 32 + lane;
  const v16bf* pw1 = pw0 + (size_t)9 * nch * 32;

  for (int tap = 0; tap < 9; ++tap) {
    const int ky = tap / 3 - 1, kx = tap % 3 - 1;
    const int iy0 = oy0 * stride + ky, ix0 = ox0 * stride + kx;
    const int iy1 = oy1 * stride + ky, ix1 = ox1 * stride + kx;
    const bool iv0 = (iy0 >= 0) & (iy0 < Hi) & (ix0 >= 0) & (ix0 < Wi);
    const bool iv1 = (iy1 >= 0) & (iy1 < Hi) & (ix1 >= 0) & (ix1 < Wi);
    const int pin0 = iy0 * Wi + ix0, pin1 = iy1 * Wi + ix1;
    const v16bf* pa0 = pw0 + (size_t)tap * nch * 32;
    const v16bf* pa1 = pw1 + (size_t)tap * nch * 32;
    __builtin_prefetch(pa0 + (size_t)nch * 32, 0, 1);   // next tap fragments

    for (int cc = 0; cc < nch; ++cc) {
      v16bf a0 = pa0[(size_t)cc * 32];                  // coalesced 32B
      v16bf a1 = pa1[(size_t)cc * 32];
      const int seg = cc * 32 + kbofs;
      const __bf16* srcB = (seg < C1) ? inA : inB;
      const int Cs  = (seg < C1) ? C1 : C2;
      const int co  = (seg < C1) ? seg : (seg - C1);
      v16bf b0 = {}, b1 = {};
      if (iv0) b0 = *reinterpret_cast<const v16bf*>(srcB + (size_t)pin0 * Cs + co);
      if (iv1) b1 = *reinterpret_cast<const v16bf*>(srcB + (size_t)pin1 * Cs + co);
      acc00 = __builtin_amdgcn_wmma_f32_16x16x32_bf16(false, a0, false, b0, (short)0, acc00, false, false);
      acc01 = __builtin_amdgcn_wmma_f32_16x16x32_bf16(false, a0, false, b1, (short)0, acc01, false, false);
      acc10 = __builtin_amdgcn_wmma_f32_16x16x32_bf16(false, a1, false, b0, (short)0, acc10, false, false);
      acc11 = __builtin_amdgcn_wmma_f32_16x16x32_bf16(false, a1, false, b1, (short)0, acc11, false, false);
    }
  }

  const int half = (lane < 16) ? 0 : 8;
  #pragma unroll
  for (int q = 0; q < 4; ++q) {
    const v8f& acc = (q == 0) ? acc00 : (q == 1) ? acc01 : (q == 2) ? acc10 : acc11;
    const int p  = (q & 1) ? p1 : p0;
    const int ocbase = (mt + (q >> 1)) * 16 + half;
    v8bf o;
    #pragma unroll
    for (int r = 0; r < 8; ++r) {
      float d = acc[r];
      if (bias) d += bias[ocbase + r];
      if (relu) d = lrelu(d);
      o[r] = (__bf16)d;
    }
    *reinterpret_cast<v8bf*>(out + (size_t)p * Cout + ocbase) = o;  // 16B store
  }
}

// ---------------------------------------------------------------------------
// Dense 32->448 per-pixel projection, 2x2 blocked (K=32 exactly).
// feat: bf16 NHWC (HW,32). out: fp32 (HW,448). bias valid for oc<441.
// ---------------------------------------------------------------------------
__global__ __launch_bounds__(128)
void dense1x1_wmma(const __bf16* __restrict__ feat, const v16bf* __restrict__ pw,
                   const float* __restrict__ bias, float* __restrict__ out,
                   int Ovalid, int Opad, int MtB, int tiles)
{
  const int lane = threadIdx.x & 31;
  const int tile = blockIdx.x * 4 + (threadIdx.x >> 5);
  if (tile >= tiles) return;
  const int mt = (tile % MtB) * 2, nt2 = tile / MtB;
  const int p0 = nt2 * 32 + (lane & 15), p1 = p0 + 16;
  const int kbofs = (lane < 16) ? 0 : 16;

  v16bf a0 = pw[(size_t)mt * 32 + lane];
  v16bf a1 = pw[(size_t)(mt + 1) * 32 + lane];
  v16bf b0 = *reinterpret_cast<const v16bf*>(feat + (size_t)p0 * 32 + kbofs);
  v16bf b1 = *reinterpret_cast<const v16bf*>(feat + (size_t)p1 * 32 + kbofs);
  v8f acc00 = {}, acc01 = {}, acc10 = {}, acc11 = {};
  acc00 = __builtin_amdgcn_wmma_f32_16x16x32_bf16(false, a0, false, b0, (short)0, acc00, false, false);
  acc01 = __builtin_amdgcn_wmma_f32_16x16x32_bf16(false, a0, false, b1, (short)0, acc01, false, false);
  acc10 = __builtin_amdgcn_wmma_f32_16x16x32_bf16(false, a1, false, b0, (short)0, acc10, false, false);
  acc11 = __builtin_amdgcn_wmma_f32_16x16x32_bf16(false, a1, false, b1, (short)0, acc11, false, false);

  const int half = (lane < 16) ? 0 : 8;
  #pragma unroll
  for (int q = 0; q < 4; ++q) {
    const v8f& acc = (q == 0) ? acc00 : (q == 1) ? acc01 : (q == 2) ? acc10 : acc11;
    const int p  = (q & 1) ? p1 : p0;
    const int ocbase = (mt + (q >> 1)) * 16 + half;
    v8f o;
    #pragma unroll
    for (int r = 0; r < 8; ++r) {
      const int oc = ocbase + r;
      o[r] = acc[r] + ((oc < Ovalid) ? bias[oc] : 0.f);
    }
    *reinterpret_cast<v8f*>(out + (size_t)p * Opad + ocbase) = o;   // 32B store
  }
}

// ---------------------------------------------------------------------------
// Bilinear 2x upsample, bf16 NHWC, half-pixel centers, edge clamp.
// ---------------------------------------------------------------------------
__global__ void upsample2x(const __bf16* __restrict__ in, __bf16* __restrict__ out,
                           int C, int H, int W)
{
  const int Ho = 2 * H, Wo = 2 * W;
  const int total = C * Ho * Wo;
  int idx = blockIdx.x * 256 + threadIdx.x;
  if (idx >= total) return;
  const int c = idx % C;
  int t = idx / C;
  const int ox = t % Wo, oy = t / Wo;

  float sy = (oy + 0.5f) * 0.5f - 0.5f;
  float sx = (ox + 0.5f) * 0.5f - 0.5f;
  int y0 = (int)floorf(sy), x0 = (int)floorf(sx);
  float fy = sy - (float)y0, fx = sx - (float)x0;
  int y0c = min(max(y0, 0), H - 1), y1c = min(max(y0 + 1, 0), H - 1);
  int x0c = min(max(x0, 0), W - 1), x1c = min(max(x0 + 1, 0), W - 1);
  float v00 = (float)in[(size_t)(y0c * W + x0c) * C + c];
  float v01 = (float)in[(size_t)(y0c * W + x1c) * C + c];
  float v10 = (float)in[(size_t)(y1c * W + x0c) * C + c];
  float v11 = (float)in[(size_t)(y1c * W + x1c) * C + c];
  float v0 = v00 + fx * (v01 - v00);
  float v1 = v10 + fx * (v11 - v10);
  out[idx] = (__bf16)(v0 + fy * (v1 - v0));
}

// ---------------------------------------------------------------------------
// Per-pixel softmax over 441 of 448 packed channels; one wave per pixel,
// contiguous row reads + wave shuffle reductions.
// ---------------------------------------------------------------------------
__global__ void softmax_row(float* __restrict__ kern, int HW)
{
  const int lane = threadIdx.x & 31;
  const int p = blockIdx.x * 8 + (threadIdx.x >> 5);
  if (p >= HW) return;
  float* row = kern + (size_t)p * 448;

  float v[14];
  float m = -3.4e38f;
  #pragma unroll
  for (int i = 0; i < 14; ++i) {
    const int c = i * 32 + lane;
    v[i] = (c < 441) ? row[c] : -3.4e38f;
    m = fmaxf(m, v[i]);
  }
  #pragma unroll
  for (int d = 16; d > 0; d >>= 1) m = fmaxf(m, __shfl_xor(m, d, 32));
  float s = 0.f;
  #pragma unroll
  for (int i = 0; i < 14; ++i) { v[i] = __expf(v[i] - m); s += v[i]; }
  #pragma unroll
  for (int d = 16; d > 0; d >>= 1) s += __shfl_xor(s, d, 32);
  const float inv = 1.f / s;
  #pragma unroll
  for (int i = 0; i < 14; ++i) {
    const int c = i * 32 + lane;
    if (c < 441) row[c] = v[i] * inv;
  }
}

// ---------------------------------------------------------------------------
// Fused 21x21 per-pixel blur (reflect pad) + 4x4 avg pool.
// img: fp32 NCHW (3,256,256). kern: fp32 (65536,448). out: (3,64,64).
// ---------------------------------------------------------------------------
__device__ __forceinline__ int refl(int i, int n) {
  if (i < 0) i = -i;
  if (i >= n) i = 2 * n - 2 - i;
  return i;
}

__global__ void blurpool(const float* __restrict__ img, const float* __restrict__ kern,
                         float* __restrict__ outp)
{
  int idx = blockIdx.x * 256 + threadIdx.x;
  if (idx >= 3 * 64 * 64) return;
  const int ox = idx % 64;
  int t = idx / 64;
  const int oy = t % 64, c = t / 64;
  const float* ic = img + (size_t)c * 65536;

  float acc = 0.f;
  for (int sy = 0; sy < 4; ++sy) {
    for (int sx = 0; sx < 4; ++sx) {
      const int y = oy * 4 + sy, x = ox * 4 + sx;
      const float* kp = kern + (size_t)(y * 256 + x) * 448;
      for (int ky = 0; ky < 21; ++ky) {
        const int yy = refl(y + ky - 10, 256);
        const float* irow = ic + yy * 256;
        #pragma unroll 7
        for (int kx = 0; kx < 21; ++kx) {
          const int xx = refl(x + kx - 10, 256);
          acc += kp[ky * 21 + kx] * irow[xx];
        }
      }
    }
  }
  outp[idx] = acc * (1.f / 16.f);
}

// ---------------------------------------------------------------------------
// Noise tail: tanh(n2(nbuf)) * |alpha|, 4x pool, += img_noise.
// ---------------------------------------------------------------------------
__global__ void noisepool(const __bf16* __restrict__ nbuf, const float* __restrict__ n2w,
                          const float* __restrict__ n2b, const float* __restrict__ alpha,
                          float* __restrict__ imgn)
{
  int idx = blockIdx.x * 256 + threadIdx.x;
  if (idx >= 3 * 64 * 64) return;
  const int ox = idx % 64;
  int t = idx / 64;
  const int oy = t % 64, c = t / 64;
  const float a = fabsf(alpha[0]);

  float s = 0.f;
  for (int sy = 0; sy < 4; ++sy) {
    for (int sx = 0; sx < 4; ++sx) {
      const int hr = (oy * 4 + sy) * 256 + (ox * 4 + sx);
      const __bf16* np = nbuf + (size_t)hr * 32;
      float v = n2b[c];
      #pragma unroll
      for (int ch = 0; ch < 32; ++ch) v += (float)np[ch] * n2w[c * 32 + ch];
      s += tanhf(v) * a;
    }
  }
  imgn[idx] += s * (1.f / 16.f);
}

// ---------------------------------------------------------------------------
// Per-channel mean of bf16 NHWC feat -> z[32].
// ---------------------------------------------------------------------------
__global__ void chanmean(const __bf16* __restrict__ in, float* __restrict__ z, int HW)
{
  __shared__ float sm[256];
  const int c = blockIdx.x;
  float s = 0.f;
  for (int i = threadIdx.x; i < HW; i += 256) s += (float)in[(size_t)i * 32 + c];
  sm[threadIdx.x] = s;
  __syncthreads();
  for (int k = 128; k > 0; k >>= 1) {
    if (threadIdx.x < k) sm[threadIdx.x] += sm[threadIdx.x + k];
    __syncthreads();
  }
  if (threadIdx.x == 0) z[c] = sm[0] / (float)HW;
}

// ---------------------------------------------------------------------------
// q-MLP: 32 -> 128 (lrelu) -> 64.
// ---------------------------------------------------------------------------
__global__ void qmlp(const float* __restrict__ z, const float* __restrict__ q1w,
                     const float* __restrict__ q1b, const float* __restrict__ q2w,
                     const float* __restrict__ q2b, float* __restrict__ qout)
{
  __shared__ float t1[128];
  const int t = threadIdx.x;
  if (t < 128) {
    float a = q1b[t];
    #pragma unroll
    for (int c = 0; c < 32; ++c) a += z[c] * q1w[t * 32 + c];
    t1[t] = lrelu(a);
  }
  __syncthreads();
  if (t < 64) {
    float a = q2b[t];
    for (int c = 0; c < 128; ++c) a += t1[c] * q2w[t * 128 + c];
    qout[t] = a;
  }
}

// ---------------------------------------------------------------------------
// "diff JPEG": ycbcr scale by Q/(Q+eps) per 8x8 position, clamps, round trip.
// ---------------------------------------------------------------------------
__global__ void jpegk(const float* __restrict__ imgn, const float* __restrict__ qv,
                      float* __restrict__ out)
{
  const int HW = 64 * 64;
  int p = blockIdx.x * 256 + threadIdx.x;
  if (p >= HW) return;
  const int y = p / 64, x = p % 64;
  const int pos = (y & 7) * 8 + (x & 7);
  float Q = 1.f / (1.f + __expf(-qv[pos])) * 255.f;
  Q = fminf(fmaxf(Q, 1.f), 255.f);
  const float s = Q / (Q + 1e-4f);

  float r = imgn[0 * HW + p] * 255.f;
  float g = imgn[1 * HW + p] * 255.f;
  float b = imgn[2 * HW + p] * 255.f;
  float Y  = (0.299f * r + 0.587f * g + 0.114f * b) * (1.f / 255.f);
  float Cb = (-0.1687f * r - 0.3313f * g + 0.5f * b + 128.f) * (1.f / 255.f);
  float Cr = (0.5f * r - 0.4187f * g - 0.0813f * b + 128.f) * (1.f / 255.f);
  Y *= s; Cb *= s; Cr *= s;
  Y  = fminf(fmaxf(Y , 0.f), 1.f) * 255.f;
  Cb = fminf(fmaxf(Cb, 0.f), 1.f) * 255.f;
  Cr = fminf(fmaxf(Cr, 0.f), 1.f) * 255.f;
  float r2 = Y + 1.402f * (Cr - 128.f);
  float g2 = Y - 0.344136f * (Cb - 128.f) - 0.714136f * (Cr - 128.f);
  float b2 = Y + 1.772f * (Cb - 128.f);
  out[0 * HW + p] = fminf(fmaxf(fminf(fmaxf(r2, 0.f), 255.f) * (1.f / 255.f), 0.f), 1.f);
  out[1 * HW + p] = fminf(fmaxf(fminf(fmaxf(g2, 0.f), 255.f) * (1.f / 255.f), 0.f), 1.f);
  out[2 * HW + p] = fminf(fmaxf(fminf(fmaxf(b2, 0.f), 255.f) * (1.f / 255.f), 0.f), 1.f);
}

// ---------------------------------------------------------------------------
// Host side
// ---------------------------------------------------------------------------
static inline __bf16* pack_layer(hipStream_t st, const float* wt, __bf16* cur,
                                 int Cout, int CinPad, int CinSrc, int ntap)
{
  const int Mt = (Cout + 15) / 16;
  const int nch = CinPad >> 5;
  const size_t total = (size_t)Mt * ntap * nch * 512;
  packw<<<(int)((total + 255) / 256), 256, 0, st>>>(wt, cur, Cout, CinSrc, CinPad,
                                                   ntap, (int)total);
  return cur + total;
}

static inline void launch_conv(hipStream_t st, const __bf16* inA, const __bf16* inB,
                               int C1, int C2, const __bf16* pw, const float* bias,
                               __bf16* out, int Cout, int Hi, int stride, int relu)
{
  const int Ho = Hi / stride;
  const int MtB = Cout / 32;                   // 2x2 blocked: Cout%32==0
  const int NtB = (Ho * Ho) / 32;              // (Ho*Wo)%32==0
  const int tiles = MtB * NtB;
  conv3x3_wmma<<<(tiles + 3) / 4, 128, 0, st>>>(
      inA, inB ? inB : inA, inB ? C1 : (C1 + C2), inB ? C2 : 0,
      (const v16bf*)pw, bias, out, Cout, Hi, Hi, Ho, stride, relu, MtB, tiles);
}

// byte offsets into d_ws (all 4KB aligned)
static constexpr size_t B_IN0  = 0;                          // 65536*32*2
static constexpr size_t B_X1   = B_IN0  + 4194304;           // 65536*32*2
static constexpr size_t B_X2   = B_X1   + 4194304;           // 16384*64*2
static constexpr size_t B_X3   = B_X2   + 2097152;           // 4096*128*2
static constexpr size_t B_X4   = B_X3   + 1048576;           // 1024*256*2
static constexpr size_t B_X5   = B_X4   + 524288;            // 256*512*2
static constexpr size_t B_FEAT = B_X5   + 262144;            // 65536*32*2
static constexpr size_t B_NBUF = B_FEAT + 4194304;           // 65536*32*2
static constexpr size_t B_KERN = B_NBUF + 4194304;           // 65536*448*4
static constexpr size_t B_IMGN = B_KERN + 117440512;         // 3*4096*4
static constexpr size_t B_MISC = B_IMGN + 49152;             // z@0, qv@1024
static constexpr size_t B_TA   = B_MISC + 4096;              // 8MB scratch
static constexpr size_t B_TB   = B_TA   + 8388608;           // 8MB scratch
static constexpr size_t B_PACK = B_TB   + 8388608;           // ~18MB packed wts

extern "C" void kernel_launch(void* const* d_in, const int* in_sizes, int n_in,
                              void* d_out, int out_size, void* d_ws, size_t ws_size,
                              hipStream_t stream)
{
  const float* img  = (const float*)d_in[0];
  const float* W[28];
  for (int i = 0; i < 27; ++i) W[i] = (const float*)d_in[1 + i];
  // W: 0 inc1,1 inc2,2 d1s,3 d1c1,4 d1c2,5 d2s,6 d2c1,7 d2c2,8 d3s,9 d3c1,
  //   10 d3c2,11 d4s,12 d4c1,13 d4c2,14 u1c1,15 u1c2,16 u2c1,17 u2c2,
  //   18 u3c1,19 u3c2,20 u4c1,21 u4c2,22 kw,23 n1w,24 n2w,25 q1w,26 q2w
  const float* kb   = (const float*)d_in[28];
  const float* n1b  = (const float*)d_in[29];
  const float* n2b  = (const float*)d_in[30];
  const float* q1b  = (const float*)d_in[31];
  const float* q2b  = (const float*)d_in[32];
  const float* alpha= (const float*)d_in[33];

  char* base = (char*)d_ws;
  __bf16* in0  = (__bf16*)(base + B_IN0);
  __bf16* x1   = (__bf16*)(base + B_X1);
  __bf16* x2   = (__bf16*)(base + B_X2);
  __bf16* x3   = (__bf16*)(base + B_X3);
  __bf16* x4   = (__bf16*)(base + B_X4);
  __bf16* x5   = (__bf16*)(base + B_X5);
  __bf16* feat = (__bf16*)(base + B_FEAT);
  __bf16* nbuf = (__bf16*)(base + B_NBUF);
  float*  kern = (float*)(base + B_KERN);
  float*  imgn = (float*)(base + B_IMGN);
  float*  z    = (float*)(base + B_MISC);
  float*  qv   = (float*)(base + B_MISC + 1024);
  __bf16* tA   = (__bf16*)(base + B_TA);
  __bf16* tB   = (__bf16*)(base + B_TB);
  float*  outp = (float*)d_out;

  // ---- pack all weights into WMMA A-fragment layout ----
  __bf16* cur = (__bf16*)(base + B_PACK);
  __bf16* pwv[24];
  static const int convCout[23] = {32,32,32,64,64,64,128,128,128,256,256,256,512,512,
                                   256,256,128,128,64,64,32,32,32};
  static const int convCin [23] = {32,32,32,32,64,64,64,128,128,128,256,256,256,512,
                                   768,256,384,128,192,64,96,32,32};
  // conv index -> W index: 0..21 -> W[0..21], 22 (n1w) -> W[23]
  for (int i = 0; i < 23; ++i) {
    const float* src = (i == 22) ? W[23] : W[i];
    const int cinSrc = (i == 0) ? 3 : convCin[i];
    pwv[i] = cur;
    cur = pack_layer(stream, src, cur, convCout[i], convCin[i], cinSrc, 9);
  }
  pwv[23] = cur;                                    // kw (441,32), ntap=1
  cur = pack_layer(stream, W[22], cur, 441, 32, 32, 1);

  // ---- input pad/convert ----
  img_to_bf16<<<(65536 * 32 + 255) / 256, 256, 0, stream>>>(img, in0);

  // ---- UNet encoder ----
  launch_conv(stream, in0, nullptr, 32, 0, pwv[0],  nullptr, tA, 32, 256, 1, 1);
  launch_conv(stream, tA,  nullptr, 32, 0, pwv[1],  nullptr, x1, 32, 256, 1, 1);

  launch_conv(stream, x1,  nullptr, 32, 0, pwv[2],  nullptr, tA, 32, 256, 2, 1);
  launch_conv(stream, tA,  nullptr, 32, 0, pwv[3],  nullptr, tB, 64, 128, 1, 1);
  launch_conv(stream, tB,  nullptr, 64, 0, pwv[4],  nullptr, x2, 64, 128, 1, 1);

  launch_conv(stream, x2,  nullptr, 64, 0, pwv[5],  nullptr, tA, 64, 128, 2, 1);
  launch_conv(stream, tA,  nullptr, 64, 0, pwv[6],  nullptr, tB, 128, 64, 1, 1);
  launch_conv(stream, tB,  nullptr, 128,0, pwv[7],  nullptr, x3, 128, 64, 1, 1);

  launch_conv(stream, x3,  nullptr, 128,0, pwv[8],  nullptr, tA, 128, 64, 2, 1);
  launch_conv(stream, tA,  nullptr, 128,0, pwv[9],  nullptr, tB, 256, 32, 1, 1);
  launch_conv(stream, tB,  nullptr, 256,0, pwv[10], nullptr, x4, 256, 32, 1, 1);

  launch_conv(stream, x4,  nullptr, 256,0, pwv[11], nullptr, tA, 256, 32, 2, 1);
  launch_conv(stream, tA,  nullptr, 256,0, pwv[12], nullptr, tB, 512, 16, 1, 1);
  launch_conv(stream, tB,  nullptr, 512,0, pwv[13], nullptr, x5, 512, 16, 1, 1);

  // ---- UNet decoder ----
  upsample2x<<<(512*32*32 + 255)/256, 256, 0, stream>>>(x5, tA, 512, 16, 16);
  launch_conv(stream, x4, tA, 256, 512, pwv[14], nullptr, tB, 256, 32, 1, 1);
  launch_conv(stream, tB, nullptr, 256, 0, pwv[15], nullptr, tA, 256, 32, 1, 1);

  upsample2x<<<(256*64*64 + 255)/256, 256, 0, stream>>>(tA, tB, 256, 32, 32);
  launch_conv(stream, x3, tB, 128, 256, pwv[16], nullptr, tA, 128, 64, 1, 1);
  launch_conv(stream, tA, nullptr, 128, 0, pwv[17], nullptr, tB, 128, 64, 1, 1);

  upsample2x<<<(128*128*128 + 255)/256, 256, 0, stream>>>(tB, tA, 128, 64, 64);
  launch_conv(stream, x2, tA, 64, 128, pwv[18], nullptr, tB, 64, 128, 1, 1);
  launch_conv(stream, tB, nullptr, 64, 0, pwv[19], nullptr, tA, 64, 128, 1, 1);

  upsample2x<<<(64*256*256 + 255)/256, 256, 0, stream>>>(tA, tB, 64, 128, 128);
  launch_conv(stream, x1, tB, 32, 64, pwv[20], nullptr, tA, 32, 256, 1, 1);
  launch_conv(stream, tA, nullptr, 32, 0, pwv[21], nullptr, feat, 32, 256, 1, 1);

  const int HW = 65536;

  // ---- kernel-prediction head: klog (WMMA) + per-pixel softmax ----
  { const int MtB = 14, tiles = MtB * (HW / 32);
    dense1x1_wmma<<<(tiles + 3) / 4, 128, 0, stream>>>(
        feat, (const v16bf*)pwv[23], kb, kern, 441, 448, MtB, tiles); }
  softmax_row<<<(HW + 7) / 8, 256, 0, stream>>>(kern, HW);

  // ---- noise head: conv3x3 + bias + lrelu ----
  launch_conv(stream, feat, nullptr, 32, 0, pwv[22], n1b, nbuf, 32, 256, 1, 1);

  // ---- q head ----
  chanmean<<<32, 256, 0, stream>>>(feat, z, HW);
  qmlp<<<1, 128, 0, stream>>>(z, W[25], q1b, W[26], q2b, qv);

  // ---- blur+pool, noise+pool ----
  blurpool <<<(3*64*64 + 255)/256, 256, 0, stream>>>(img, kern, imgn);
  noisepool<<<(3*64*64 + 255)/256, 256, 0, stream>>>(nbuf, W[24], n2b, alpha, imgn);

  // ---- diff-JPEG + final clip ----
  jpegk<<<(64*64 + 255)/256, 256, 0, stream>>>(imgn, qv, outp);

  (void)in_sizes; (void)n_in; (void)out_size; (void)ws_size;
}
